// SequentialRelative_51419348467969
// MI455X (gfx1250) — matrix operand order repaired and checked by reading, not compiled, not run on hardware
//
#include <hip/hip_runtime.h>

// ---------------------------------------------------------------------------
// CDNA5 (gfx1250) wave32 WMMA transformer encoder-decoder.
// All GEMMs run through v_wmma_f32_16x16x32_bf16 (fp32 -> bf16 at LDS stage,
// fp32 accumulate).  Compute-bound workload (~1 TFLOP vs ~26us of HBM traffic).
// ---------------------------------------------------------------------------

typedef __bf16 bf16;
typedef __bf16 v16bf __attribute__((ext_vector_type(16)));
typedef float  v8f   __attribute__((ext_vector_type(8)));

union Frag16 { v16bf v; unsigned u[8]; };

__device__ __forceinline__ v8f wmma_bf16(v16bf a, v16bf b, v8f c) {
  // D = A(16x32 bf16) * B(32x16 bf16) + C(16x16 f32)
  return __builtin_amdgcn_wmma_f32_16x16x32_bf16(
      /*neg_a=*/false, a, /*neg_b=*/false, b,
      /*c_mod=*/(short)0, c, /*reuse_a=*/false, /*reuse_b=*/false);
}

// ---------------------------------------------------------------------------
// Generic GEMM: C[M,N] = A[M,K] x B[K,N]   (or x B^T with transB, B stored [N,K])
// fp32 in / fp32 out, bf16 WMMA inner product.  Block tile 128x128, K-step 32.
// 8 waves arranged 2x4; each wave: 64x32 sub-tile = 4x2 WMMA accumulators.
// blockIdx.z batching via strideAz/Bz/Cz.
// ---------------------------------------------------------------------------
template <bool RELU>
__global__ __launch_bounds__(256) void gemm_bf16_wmma(
    const float* __restrict__ A, int lda, long long sAz,
    const float* __restrict__ Bm, int ldb, long long sBz, int transB,
    float* __restrict__ C, int ldc, long long sCz,
    int M, int N, int K) {
  __shared__ bf16 As[128 * 32];  // [m][k]
  __shared__ bf16 Bs[128 * 32];  // [n][k]  (k-major per column for packed pairs)

  const int tid = threadIdx.x;
  const int lane = tid & 31;
  const int wave = tid >> 5;
  const int wr = wave >> 2;   // 0..1
  const int wc = wave & 3;    // 0..3
  const int z = blockIdx.z;
  A  += (size_t)z * sAz;
  Bm += (size_t)z * sBz;
  C  += (size_t)z * sCz;
  const int m0 = blockIdx.y * 128;
  const int n0 = blockIdx.x * 128;

  const v8f vzero = {0.f, 0.f, 0.f, 0.f, 0.f, 0.f, 0.f, 0.f};
  v8f acc[4][2];
#pragma unroll
  for (int i = 0; i < 4; ++i)
#pragma unroll
    for (int j = 0; j < 2; ++j) acc[i][j] = vzero;

  // Operand lane mapping used here (wave32, contiguous k = 16h + s, same for A and B):
  const int rA   = lane & 15;             // A row within 16
  const int kbA  = (lane & 16) ? 16 : 0;  // A K-base: lanes16-31 hold K 16..31
  const int colB = lane & 15;             // B column
  const int kbB  = (lane & 16) ? 16 : 0;  // B K-base

  for (int k0 = 0; k0 < K; k0 += 32) {
    // ---- stage A tile (fp32 -> bf16) ----
#pragma unroll 4
    for (int e = tid; e < 128 * 32; e += 256) {
      int r = e >> 5, k = e & 31;
      int gm = m0 + r;
      float v = (gm < M) ? A[(size_t)gm * lda + (k0 + k)] : 0.0f;
      As[e] = (bf16)v;
    }
    // ---- stage B tile into Bs[n][k] ----
    if (transB) {
#pragma unroll 4
      for (int e = tid; e < 128 * 32; e += 256) {
        int c = e >> 5, k = e & 31;
        int gn = n0 + c;
        float v = (gn < N) ? Bm[(size_t)gn * ldb + (k0 + k)] : 0.0f;
        Bs[c * 32 + k] = (bf16)v;
      }
    } else {
#pragma unroll 4
      for (int e = tid; e < 128 * 32; e += 256) {
        int k = e >> 7, c = e & 127;
        int gn = n0 + c;
        float v = (gn < N) ? Bm[(size_t)(k0 + k) * ldb + gn] : 0.0f;
        Bs[c * 32 + k] = (bf16)v;
      }
    }
    // prefetch next weight tile (global_prefetch_b8 on gfx1250)
    if (k0 + 32 < K && !transB)
      __builtin_prefetch(&Bm[(size_t)(k0 + 32 + (tid >> 7)) * ldb + n0 + (tid & 127)], 0, 1);
    __syncthreads();

    const unsigned* As32 = (const unsigned*)As;
    const unsigned* Bs32 = (const unsigned*)Bs;

    Frag16 bf[2];
#pragma unroll
    for (int nt = 0; nt < 2; ++nt) {
      int c = wc * 32 + nt * 16 + colB;
      int base = c * 16 + (kbB >> 1);
#pragma unroll
      for (int i = 0; i < 8; ++i) bf[nt].u[i] = Bs32[base + i];
    }
#pragma unroll
    for (int mt = 0; mt < 4; ++mt) {
      int r = wr * 64 + mt * 16 + rA;
      int rb = r * 16 + (kbA >> 1);
      Frag16 af;
#pragma unroll
      for (int i = 0; i < 4; ++i) af.u[i] = As32[rb + i];          // K kbA..kbA+7
#pragma unroll
      for (int i = 0; i < 4; ++i) af.u[4 + i] = As32[rb + 4 + i];  // K kbA+8..+15
#pragma unroll
      for (int nt = 0; nt < 2; ++nt)
        acc[mt][nt] = wmma_bf16(af.v, bf[nt].v, acc[mt][nt]);
    }
    __syncthreads();
  }

  const int rowOff = (lane & 16) ? 8 : 0;
#pragma unroll
  for (int mt = 0; mt < 4; ++mt)
#pragma unroll
    for (int nt = 0; nt < 2; ++nt)
#pragma unroll
      for (int g = 0; g < 8; ++g) {
        int m = m0 + wr * 64 + mt * 16 + g + rowOff;
        int n = n0 + wc * 32 + nt * 16 + colB;
        if (m < M && n < N) {
          float v = acc[mt][nt][g];
          if (RELU) v = v > 0.f ? v : 0.f;
          C[(size_t)m * ldc + n] = v;
        }
      }
}

// ---------------------------------------------------------------------------
// Attention scores: scores[b,h,i,j] = (q.k + qR[b,i,h,rel(i,j)]) / sqrt(hd),
// masked.  Tile 64(i) x 128(j), waves 2x4, each wave 32x32 (2x2 WMMA accums).
// ---------------------------------------------------------------------------
__global__ __launch_bounds__(256) void attn_scores_wmma(
    const float* __restrict__ q, const float* __restrict__ kk,
    const float* __restrict__ qR, const float* __restrict__ mask,
    float* __restrict__ scores, int b, int I, int J, int D, int H, int hd) {
  __shared__ bf16 Qs[64 * 32];   // [i][k]
  __shared__ bf16 Ks[128 * 32];  // [j][k]

  const int tid = threadIdx.x;
  const int lane = tid & 31;
  const int wave = tid >> 5;
  const int wr = wave >> 2;  // 0..1 (i)
  const int wc = wave & 3;   // 0..3 (j)
  const int h = blockIdx.z;
  const int i0 = blockIdx.y * 64;
  const int j0 = blockIdx.x * 128;

  const float* qp = q  + (size_t)b * I * D + (size_t)h * hd;
  const float* kp = kk + (size_t)b * J * D + (size_t)h * hd;

  const v8f vzero = {0.f, 0.f, 0.f, 0.f, 0.f, 0.f, 0.f, 0.f};
  v8f acc[2][2];
#pragma unroll
  for (int i = 0; i < 2; ++i)
#pragma unroll
    for (int j = 0; j < 2; ++j) acc[i][j] = vzero;

  const int rA   = lane & 15;
  const int kbA  = (lane & 16) ? 16 : 0;
  const int colB = lane & 15;
  const int kbB  = (lane & 16) ? 16 : 0;

  for (int k0 = 0; k0 < hd; k0 += 32) {
#pragma unroll 2
    for (int e = tid; e < 64 * 32; e += 256) {
      int r = e >> 5, kd = e & 31;
      Qs[e] = (bf16)qp[(size_t)(i0 + r) * D + k0 + kd];
    }
#pragma unroll 4
    for (int e = tid; e < 128 * 32; e += 256) {
      int r = e >> 5, kd = e & 31;
      Ks[e] = (bf16)kp[(size_t)(j0 + r) * D + k0 + kd];
    }
    __syncthreads();

    const unsigned* Qs32 = (const unsigned*)Qs;
    const unsigned* Ks32 = (const unsigned*)Ks;
    Frag16 bf[2];
#pragma unroll
    for (int nt = 0; nt < 2; ++nt) {
      int c = wc * 32 + nt * 16 + colB;
      int base = c * 16 + (kbB >> 1);
#pragma unroll
      for (int i = 0; i < 8; ++i) bf[nt].u[i] = Ks32[base + i];
    }
#pragma unroll
    for (int mt = 0; mt < 2; ++mt) {
      int r = wr * 32 + mt * 16 + rA;
      int rb = r * 16 + (kbA >> 1);
      Frag16 af;
#pragma unroll
      for (int i = 0; i < 4; ++i) af.u[i] = Qs32[rb + i];
#pragma unroll
      for (int i = 0; i < 4; ++i) af.u[4 + i] = Qs32[rb + 4 + i];
#pragma unroll
      for (int nt = 0; nt < 2; ++nt)
        acc[mt][nt] = wmma_bf16(af.v, bf[nt].v, acc[mt][nt]);
    }
    __syncthreads();
  }

  const float scale = 0.08838834764831845f;  // 1/sqrt(128)
  const int rowOff = (lane & 16) ? 8 : 0;
#pragma unroll
  for (int mt = 0; mt < 2; ++mt)
#pragma unroll
    for (int nt = 0; nt < 2; ++nt)
#pragma unroll
      for (int g = 0; g < 8; ++g) {
        int i = i0 + wr * 32 + mt * 16 + g + rowOff;
        int j = j0 + wc * 32 + nt * 16 + colB;
        int d = i - j;
        d = d < -64 ? -64 : (d > 64 ? 64 : d);
        float bias = qR[(((size_t)b * I + i) * H + h) * 129 + (d + 64)];
        float s = (acc[mt][nt][g] + bias) * scale;
        float m = mask[((size_t)b * I + i) * J + j];
        s = (m > 0.f) ? s : -1e9f;
        scores[(((size_t)(b * H + h) * I) + i) * (size_t)J + j] = s;
      }
}

// ---------------------------------------------------------------------------
// Row softmax (in place), one block per row.
// ---------------------------------------------------------------------------
__global__ __launch_bounds__(256) void softmax_rows(float* __restrict__ s, int J) {
  __shared__ float red[256];
  float* p = s + (size_t)blockIdx.x * J;
  const int tid = threadIdx.x;
  float mx = -1e30f;
  for (int j = tid; j < J; j += 256) mx = fmaxf(mx, p[j]);
  red[tid] = mx;
  __syncthreads();
  for (int o = 128; o > 0; o >>= 1) {
    if (tid < o) red[tid] = fmaxf(red[tid], red[tid + o]);
    __syncthreads();
  }
  mx = red[0];
  __syncthreads();
  float sum = 0.f;
  for (int j = tid; j < J; j += 256) {
    float e = __expf(p[j] - mx);
    p[j] = e;
    sum += e;
  }
  red[tid] = sum;
  __syncthreads();
  for (int o = 128; o > 0; o >>= 1) {
    if (tid < o) red[tid] += red[tid + o];
    __syncthreads();
  }
  float inv = 1.0f / red[0];
  for (int j = tid; j < J; j += 256) p[j] *= inv;
}

// ---------------------------------------------------------------------------
// h[row,:] = LayerNorm(h[row,:] + x[row,:]) * g + b   (D = 1024, in place on h)
// ---------------------------------------------------------------------------
__global__ __launch_bounds__(256) void add_layernorm(
    float* __restrict__ h, const float* __restrict__ x,
    const float* __restrict__ g, const float* __restrict__ bta, int D) {
  __shared__ float red[256];
  float* hp = h + (size_t)blockIdx.x * D;
  const float* xp = x + (size_t)blockIdx.x * D;
  const int tid = threadIdx.x;
  float v[4];
  float s = 0.f;
#pragma unroll
  for (int c = 0; c < 4; ++c) {
    int d = tid + c * 256;
    v[c] = hp[d] + xp[d];
    s += v[c];
  }
  red[tid] = s;
  __syncthreads();
  for (int o = 128; o > 0; o >>= 1) {
    if (tid < o) red[tid] += red[tid + o];
    __syncthreads();
  }
  float mu = red[0] * (1.0f / 1024.0f);
  __syncthreads();
  float vs = 0.f;
#pragma unroll
  for (int c = 0; c < 4; ++c) {
    float dlt = v[c] - mu;
    vs += dlt * dlt;
  }
  red[tid] = vs;
  __syncthreads();
  for (int o = 128; o > 0; o >>= 1) {
    if (tid < o) red[tid] += red[tid + o];
    __syncthreads();
  }
  float rs = rsqrtf(red[0] * (1.0f / 1024.0f) + 1e-5f);
#pragma unroll
  for (int c = 0; c < 4; ++c) {
    int d = tid + c * 256;
    hp[d] = (v[c] - mu) * rs * g[d] + bta[d];
  }
}

// ---------------------------------------------------------------------------
// Embedding gather: out[t,:] = emb[ids[t],:] * sqrt(D)
// ---------------------------------------------------------------------------
__global__ __launch_bounds__(256) void embed_scale(
    const int* __restrict__ ids, const float* __restrict__ emb,
    float* __restrict__ out, int n, int D, float scale) {
  size_t idx = (size_t)blockIdx.x * 256 + threadIdx.x;
  size_t total = (size_t)n * D;
  if (idx < total) {
    size_t t = idx / (unsigned)D;
    size_t d = idx - t * D;
    out[idx] = emb[(size_t)ids[t] * D + d] * scale;
  }
}

// ---------------------------------------------------------------------------
// Host-side orchestration
// ---------------------------------------------------------------------------
static void gemm(const float* A, int lda, long long sAz,
                 const float* B, int ldb, long long sBz, int transB,
                 float* C, int ldc, long long sCz,
                 int M, int N, int K, int Z, bool relu, hipStream_t s) {
  dim3 grid((N + 127) / 128, (M + 127) / 128, Z);
  if (relu)
    gemm_bf16_wmma<true><<<grid, 256, 0, s>>>(A, lda, sAz, B, ldb, sBz, transB,
                                              C, ldc, sCz, M, N, K);
  else
    gemm_bf16_wmma<false><<<grid, 256, 0, s>>>(A, lda, sAz, B, ldb, sBz, transB,
                                               C, ldc, sCz, M, N, K);
}

struct WsBufs {
  float *qb, *kb, *vb, *ctx, *prj, *mid, *qR, *sc;
};

// One MHA: proj <- (softmax((q k^T + bias)/sqrt(hd)) v) Wo
static void mha_pass(const float* xq, const float* xkv, const float* mask,
                     const float* W, const float* R, const WsBufs& w,
                     int Bn, int I, int J, int D, int H, int hd, hipStream_t s) {
  const long long DD = (long long)D * D;
  // q / k / v projections
  gemm(xq,  D, 0, W + 0 * DD, D, 0, 0, w.qb, D, 0, Bn * I, D, D, 1, false, s);
  gemm(xkv, D, 0, W + 1 * DD, D, 0, 0, w.kb, D, 0, Bn * J, D, D, 1, false, s);
  gemm(xkv, D, 0, W + 2 * DD, D, 0, 0, w.vb, D, 0, Bn * J, D, D, 1, false, s);
  // qR = q (flattened [B*I*H, hd]) @ R^T  -> [B*I*H, 129]
  gemm(w.qb, hd, 0, R, hd, 0, 1, w.qR, 129, 0, Bn * I * H, 129, hd, 1, false, s);
  // scores + bias + mask
  for (int b = 0; b < Bn; ++b) {
    dim3 grid(J / 128, I / 64, H);
    attn_scores_wmma<<<grid, 256, 0, s>>>(w.qb, w.kb, w.qR, mask, w.sc,
                                          b, I, J, D, H, hd);
  }
  // softmax over J
  softmax_rows<<<(unsigned)(Bn * H * I), 256, 0, s>>>(w.sc, J);
  // ctx[b,i,h,:] = attn[b,h,i,:] @ v[b,:,h,:]
  for (int b = 0; b < Bn; ++b) {
    gemm(w.sc + (size_t)b * H * I * J, J, (long long)I * J,
         w.vb + (size_t)b * J * D, D, (long long)hd, 0,
         w.ctx + (size_t)b * I * D, D, (long long)hd,
         I, hd, J, H, false, s);
  }
  // output projection
  gemm(w.ctx, D, 0, W + 3 * DD, D, 0, 0, w.prj, D, 0, Bn * I, D, D, 1, false, s);
}

extern "C" void kernel_launch(void* const* d_in, const int* in_sizes, int n_in,
                              void* d_out, int out_size, void* d_ws, size_t ws_size,
                              hipStream_t stream) {
  const int Bn = 2, L = 1024, D = 1024, H = 8, hd = 128, V = 32000;
  const int Ne = 4, Nd = 4, FF = 4096;

  const int*   enc_ids    = (const int*)d_in[0];
  const int*   dec_ids    = (const int*)d_in[1];
  const float* enc_mask   = (const float*)d_in[2];
  const float* dec_mask   = (const float*)d_in[3];
  const float* cross_mask = (const float*)d_in[4];
  const float* emb        = (const float*)d_in[5];
  const float* rel_emb    = (const float*)d_in[6];
  const float* enc_attn   = (const float*)d_in[7];
  const float* enc_f1     = (const float*)d_in[8];
  const float* enc_f2     = (const float*)d_in[9];
  const float* enc_g      = (const float*)d_in[10];
  const float* enc_b      = (const float*)d_in[11];
  const float* dec_sattn  = (const float*)d_in[12];
  const float* dec_cattn  = (const float*)d_in[13];
  const float* dec_f1     = (const float*)d_in[14];
  const float* dec_f2     = (const float*)d_in[15];
  const float* dec_g      = (const float*)d_in[16];
  const float* dec_b      = (const float*)d_in[17];
  float* out = (float*)d_out;
  float* ws = (float*)d_ws;

  // workspace layout (floats)
  size_t o = 0;
  float* enc_h = ws + o; o += (size_t)Bn * L * D;
  float* dec_h = ws + o; o += (size_t)Bn * L * D;
  WsBufs w;
  w.qb  = ws + o; o += (size_t)Bn * L * D;
  w.kb  = ws + o; o += (size_t)Bn * L * D;
  w.vb  = ws + o; o += (size_t)Bn * L * D;
  w.ctx = ws + o; o += (size_t)Bn * L * D;
  w.prj = ws + o; o += (size_t)Bn * L * D;
  w.mid = ws + o; o += (size_t)Bn * L * FF;
  w.qR  = ws + o; o += (size_t)Bn * L * H * 129;
  w.sc  = ws + o; o += (size_t)Bn * H * L * L;
  (void)ws_size; (void)n_in; (void)in_sizes; (void)out_size;

  const float emb_scale = 32.0f;  // sqrt(1024)
  {
    size_t total = (size_t)Bn * L * D;
    unsigned nb = (unsigned)((total + 255) / 256);
    embed_scale<<<nb, 256, 0, stream>>>(enc_ids, emb, enc_h, Bn * L, D, emb_scale);
    embed_scale<<<nb, 256, 0, stream>>>(dec_ids, emb, dec_h, Bn * L, D, emb_scale);
  }

  const long long DD = (long long)D * D;
  const int rows = Bn * L;

  // ---------------- encoder ----------------
  for (int l = 0; l < Ne; ++l) {
    const float* W  = enc_attn + (size_t)l * 4 * DD;
    const float* g0 = enc_g + (size_t)l * 2 * D;
    const float* b0 = enc_b + (size_t)l * 2 * D;
    mha_pass(enc_h, enc_h, enc_mask, W, rel_emb, w, Bn, L, L, D, H, hd, stream);
    add_layernorm<<<rows, 256, 0, stream>>>(enc_h, w.prj, g0, b0, D);
    // FFN
    gemm(enc_h, D, 0, enc_f1 + (size_t)l * D * FF, FF, 0, 0,
         w.mid, FF, 0, rows, FF, D, 1, true, stream);
    gemm(w.mid, FF, 0, enc_f2 + (size_t)l * FF * D, D, 0, 0,
         w.prj, D, 0, rows, D, FF, 1, false, stream);
    add_layernorm<<<rows, 256, 0, stream>>>(enc_h, w.prj, g0 + D, b0 + D, D);
  }

  // ---------------- decoder ----------------
  for (int l = 0; l < Nd; ++l) {
    const float* Ws = dec_sattn + (size_t)l * 4 * DD;
    const float* Wc = dec_cattn + (size_t)l * 4 * DD;
    const float* g0 = dec_g + (size_t)l * 3 * D;
    const float* b0 = dec_b + (size_t)l * 3 * D;
    // masked self-attention
    mha_pass(dec_h, dec_h, dec_mask, Ws, rel_emb, w, Bn, L, L, D, H, hd, stream);
    add_layernorm<<<rows, 256, 0, stream>>>(dec_h, w.prj, g0, b0, D);
    // cross attention (kv from encoder output)
    mha_pass(dec_h, enc_h, cross_mask, Wc, rel_emb, w, Bn, L, L, D, H, hd, stream);
    add_layernorm<<<rows, 256, 0, stream>>>(dec_h, w.prj, g0 + D, b0 + D, D);
    // FFN
    gemm(dec_h, D, 0, dec_f1 + (size_t)l * D * FF, FF, 0, 0,
         w.mid, FF, 0, rows, FF, D, 1, true, stream);
    gemm(w.mid, FF, 0, dec_f2 + (size_t)l * FF * D, D, 0, 0,
         w.prj, D, 0, rows, D, FF, 1, false, stream);
    add_layernorm<<<rows, 256, 0, stream>>>(dec_h, w.prj, g0 + 2 * D, b0 + 2 * D, D);
  }

  // ---------------- logits: dec_h @ emb^T ----------------
  gemm(dec_h, D, 0, emb, D, 0, 1, out, V, 0, rows, V, D, 1, false, stream);
}
